// ipt_76287209111763
// MI455X (gfx1250) — matrix operand order repaired and checked by reading, not compiled, not run on hardware
//
#include <hip/hip_runtime.h>
#include <hip/hip_bf16.h>

typedef __attribute__((ext_vector_type(16))) _Float16 v16h;
typedef __attribute__((ext_vector_type(8)))  _Float16 v8h;
typedef __attribute__((ext_vector_type(8)))  float    v8f;

namespace {
constexpr int S_   = 256;
constexpr int B_   = 4;
constexpr int D_   = 576;
constexpr int NH_  = 12;
constexpr int HD_  = 48;
constexpr int HID_ = 2304;
constexpr int NTOK = S_ * B_;   // 1024
constexpr int BH   = B_ * NH_;  // 48
constexpr int D3   = 3 * D_;    // 1728
constexpr int HDP  = 64;        // padded head dim (48 -> 64, zero fill)
constexpr int NL_  = 4;
}

// ---------------------------------------------------------------------------
// Generic batched WMMA GEMM:  C[m,n] = sum_k A[m,k] * W[n,k]  (+bias,relu,res)
// A: f16 M x K (row-major, lda), W: f16 N x K (row-major, ldb), C: f32 (ldc)
// Each wave computes a 32 x (16*NT) output tile: 2 A-fragments reused across
// NT B-fragments -> 2*NT WMMA per 32-wide K step. K must be a multiple of 32.
// flags: 1=add bias[n], 2=relu, 4=add residual R (same layout as C)
// ---------------------------------------------------------------------------
template <int NT>
__launch_bounds__(256)
__global__ void gemm_wmma_k(const _Float16* __restrict__ A,
                            const _Float16* __restrict__ Wm,
                            const float* __restrict__ bias,
                            const float* __restrict__ R,
                            float* __restrict__ C,
                            int M, int N, int K, int lda, int ldb, int ldc,
                            long long aS, long long wS, long long cS, long long rS,
                            int flags) {
  const int mt  = M >> 5;
  const int ntb = N / (16 * NT);
  const int wave = blockIdx.x * 8 + (threadIdx.x >> 5);
  if (wave >= mt * ntb) return;  // wave-uniform exit: EXEC stays all-ones
  const int tm = wave % mt, tn = wave / mt;

  const long long bb = blockIdx.y;
  const _Float16* Ab = A + bb * aS;
  const _Float16* Wb = Wm + bb * wS;
  float* Cb = C + bb * cS;
  const float* Rb = (flags & 4) ? (R + bb * rS) : nullptr;

  const int lane = threadIdx.x & 31;
  const int l16  = lane & 15;
  const int hi   = lane >> 4;  // 0: lanes 0-15, 1: lanes 16-31

  // A operand, this code's k order (not ISA table order): lane holds rows; halves 0-15 -> K = 16*hi + e
  const _Float16* a0p = Ab + (long long)(tm * 32 + l16) * lda + hi * 16;
  const _Float16* a1p = a0p + (long long)16 * lda;
  // B operand, same k order as A: lanes 0-15 hold K=0..15, lanes 16-31 K=16..31
  const _Float16* bp[NT];
#pragma unroll
  for (int j = 0; j < NT; ++j)
    bp[j] = Wb + (long long)(tn * 16 * NT + j * 16 + l16) * ldb + hi * 16;

  v8f acc0[NT] = {};
  v8f acc1[NT] = {};
  for (int kk = 0; kk < K; kk += 32) {
    v8h al0 = *(const v8h*)(a0p + kk);
    v8h al1 = *(const v8h*)(a0p + kk + 8);
    v8h ah0 = *(const v8h*)(a1p + kk);
    v8h ah1 = *(const v8h*)(a1p + kk + 8);
    v16h av0 = __builtin_shufflevector(al0, al1, 0,1,2,3,4,5,6,7,8,9,10,11,12,13,14,15);
    v16h av1 = __builtin_shufflevector(ah0, ah1, 0,1,2,3,4,5,6,7,8,9,10,11,12,13,14,15);
#pragma unroll
    for (int j = 0; j < NT; ++j) {
      v16h bv = *(const v16h*)(bp[j] + kk);
      acc0[j] = __builtin_amdgcn_wmma_f32_16x16x32_f16(false, av0, false, bv, (short)0, acc0[j], false, false);
      acc1[j] = __builtin_amdgcn_wmma_f32_16x16x32_f16(false, av1, false, bv, (short)0, acc1[j], false, false);
    }
  }

#pragma unroll
  for (int j = 0; j < NT; ++j) {
    const int ncol = tn * 16 * NT + j * 16 + l16;
    const float bval = (flags & 1) ? bias[ncol] : 0.0f;
#pragma unroll
    for (int v = 0; v < 8; ++v) {
      const int m0 = tm * 32 + v + hi * 8;  // C/D layout: VGPR v -> M = v + 8*hi
      const int m1 = m0 + 16;
      float x0 = acc0[j][v] + bval;
      float x1 = acc1[j][v] + bval;
      if (flags & 2) { x0 = fmaxf(x0, 0.0f); x1 = fmaxf(x1, 0.0f); }
      if (flags & 4) {
        x0 += Rb[(long long)m0 * ldc + ncol];
        x1 += Rb[(long long)m1 * ldc + ncol];
      }
      Cb[(long long)m0 * ldc + ncol] = x0;
      Cb[(long long)m1 * ldc + ncol] = x1;
    }
  }
}

// ------------------------- LSQ weight quantization -------------------------
__launch_bounds__(256)
__global__ void abssum_k(const float* __restrict__ w, long long n, float* __restrict__ part) {
  __shared__ float sb[256];
  float s = 0.0f;
  for (long long i = (long long)blockIdx.x * 256 + threadIdx.x; i < n;
       i += (long long)gridDim.x * 256)
    s += fabsf(w[i]);
  sb[threadIdx.x] = s; __syncthreads();
  for (int st = 128; st > 0; st >>= 1) {
    if ((int)threadIdx.x < st) sb[threadIdx.x] += sb[threadIdx.x + st];
    __syncthreads();
  }
  if (threadIdx.x == 0) part[blockIdx.x] = sb[0];
}

__global__ void alpha_k(const float* __restrict__ part, int nb, float invn,
                        float* __restrict__ alpha) {
  if (threadIdx.x == 0) {
    float s = 0.0f;
    for (int i = 0; i < nb; ++i) s += part[i];
    *alpha = 2.0f * (s * invn) / sqrtf(7.0f);
  }
}

__launch_bounds__(256)
__global__ void quant_k(const float* __restrict__ w, const float* __restrict__ alpha,
                        _Float16* __restrict__ qw, long long n) {
  long long i = (long long)blockIdx.x * 256 + threadIdx.x;
  if (i >= n) return;
  const float a = *alpha;
  float q = w[i] / a;
  q = rintf(fminf(fmaxf(q, -8.0f), 7.0f));   // round-half-even like jnp.round
  qw[i] = (_Float16)(q * a);
}

// --------------------------- elementwise kernels ---------------------------
__launch_bounds__(256)
__global__ void patchify_k(const float* __restrict__ x, float* __restrict__ t32,
                           _Float16* __restrict__ t16) {
  int i = blockIdx.x * 256 + threadIdx.x;
  if (i >= NTOK * D_) return;
  const int d = i % D_;
  const int b = (i / D_) % B_;
  const int s = i / (D_ * B_);
  const int c = d / 9, r = d % 9, p1 = r / 3, p2 = r % 3;
  const int gy = s / 16, gx = s % 16;
  const float v = x[(((long long)b * 64 + c) * 48 + (gy * 3 + p1)) * 48 + (gx * 3 + p2)];
  t32[i] = v; t16[i] = (_Float16)v;
}

__launch_bounds__(256)
__global__ void unpatchify_k(const float* __restrict__ o, float* __restrict__ y) {
  int i = blockIdx.x * 256 + threadIdx.x;
  if (i >= B_ * 64 * 48 * 48) return;
  const int j  = i % 48;
  const int ii = (i / 48) % 48;
  const int c  = (i / (48 * 48)) % 64;
  const int b  = i / (48 * 48 * 64);
  const int gy = ii / 3, p1 = ii % 3, gx = j / 3, p2 = j % 3;
  const int s  = gy * 16 + gx;
  const int d  = c * 9 + p1 * 3 + p2;
  y[i] = o[((long long)s * B_ + b) * D_ + d];
}

// out = a + rows[(qidx)*idxScale + s*D + d]  (row matrix broadcast over batch)
__launch_bounds__(256)
__global__ void addrow_k(const float* __restrict__ a, const float* __restrict__ rows,
                         const int* __restrict__ qidx, long long idxScale,
                         float* __restrict__ out32, _Float16* __restrict__ out16, int n) {
  int i = blockIdx.x * 256 + threadIdx.x;
  if (i >= n) return;
  const int d = i % D_;
  const int s = i / (B_ * D_);
  const float* rb = rows + (qidx ? (long long)qidx[0] * idxScale : 0);
  const float v = a[i] + rb[(long long)s * D_ + d];
  if (out32) out32[i] = v;
  if (out16) out16[i] = (_Float16)v;
}

__launch_bounds__(256)
__global__ void cast16_k(const float* __restrict__ a, _Float16* __restrict__ o, int n) {
  int i = blockIdx.x * 256 + threadIdx.x;
  if (i < n) o[i] = (_Float16)a[i];
}

__launch_bounds__(256)
__global__ void copy_k(const float* __restrict__ a, float* __restrict__ o, int n) {
  int i = blockIdx.x * 256 + threadIdx.x;
  if (i < n) o[i] = a[i];
}

__launch_bounds__(256)
__global__ void layernorm_k(const float* __restrict__ x, const float* __restrict__ g,
                            const float* __restrict__ b, float* __restrict__ y32,
                            _Float16* __restrict__ y16) {
  __shared__ float sb[256];
  const int row = blockIdx.x;
  const float* xr = x + (long long)row * D_;
  float s = 0.0f, s2 = 0.0f;
  for (int i = threadIdx.x; i < D_; i += 256) { float v = xr[i]; s += v; s2 += v * v; }
  sb[threadIdx.x] = s; __syncthreads();
  for (int st = 128; st > 0; st >>= 1) {
    if ((int)threadIdx.x < st) sb[threadIdx.x] += sb[threadIdx.x + st];
    __syncthreads();
  }
  const float mean = sb[0] * (1.0f / D_);
  __syncthreads();
  sb[threadIdx.x] = s2; __syncthreads();
  for (int st = 128; st > 0; st >>= 1) {
    if ((int)threadIdx.x < st) sb[threadIdx.x] += sb[threadIdx.x + st];
    __syncthreads();
  }
  const float var = sb[0] * (1.0f / D_) - mean * mean;
  const float inv = rsqrtf(var + 1e-5f);
  for (int i = threadIdx.x; i < D_; i += 256) {
    const float v = (xr[i] - mean) * inv * g[i] + b[i];
    y32[(long long)row * D_ + i] = v;
    y16[(long long)row * D_ + i] = (_Float16)v;
  }
}

__launch_bounds__(256)
__global__ void softmax_k(const float* __restrict__ s, _Float16* __restrict__ att) {
  __shared__ float sb[256];
  const long long row = blockIdx.x;
  const float v = s[row * 256 + threadIdx.x];
  sb[threadIdx.x] = v; __syncthreads();
  for (int st = 128; st > 0; st >>= 1) {
    if ((int)threadIdx.x < st) sb[threadIdx.x] = fmaxf(sb[threadIdx.x], sb[threadIdx.x + st]);
    __syncthreads();
  }
  const float mx = sb[0]; __syncthreads();
  const float e = expf(v - mx);
  sb[threadIdx.x] = e; __syncthreads();
  for (int st = 128; st > 0; st >>= 1) {
    if ((int)threadIdx.x < st) sb[threadIdx.x] += sb[threadIdx.x + st];
    __syncthreads();
  }
  att[row * 256 + threadIdx.x] = (_Float16)(e / sb[0]);
}

// qkv (S,B,3D) f32 -> qp/kp (BH, S, 64) f16, q scaled by 1/sqrt(HD), zero pad
__launch_bounds__(256)
__global__ void repack_qk_k(const float* __restrict__ qkv, _Float16* __restrict__ qp,
                            _Float16* __restrict__ kp) {
  int i = blockIdx.x * 256 + threadIdx.x;
  if (i >= BH * S_ * HDP) return;
  const int d  = i & 63;
  const int s  = (i >> 6) & 255;
  const int bh = i >> 14;
  const int b = bh / NH_, h = bh % NH_;
  float qv = 0.0f, kv = 0.0f;
  if (d < HD_) {
    const long long base = ((long long)s * B_ + b) * D3 + h * HD_ + d;
    qv = qkv[base] * 0.14433756729740643f;  // 1/sqrt(48)
    kv = qkv[base + D_];
  }
  qp[i] = (_Float16)qv;
  kp[i] = (_Float16)kv;
}

// qkv (S,B,3D) f32 -> vt (BH, HD, S) f16  (V transposed so it can be a W operand)
__launch_bounds__(256)
__global__ void repack_vt_k(const float* __restrict__ qkv, _Float16* __restrict__ vt) {
  int i = blockIdx.x * 256 + threadIdx.x;
  if (i >= BH * HD_ * S_) return;
  const int s  = i & 255;
  const int d  = (i >> 8) % HD_;
  const int bh = i / (HD_ * S_);
  const int b = bh / NH_, h = bh % NH_;
  vt[i] = (_Float16)qkv[((long long)s * B_ + b) * D3 + 2 * D_ + h * HD_ + d];
}

// ---------------------------------------------------------------------------
extern "C" void kernel_launch(void* const* d_in, const int* in_sizes, int n_in,
                              void* d_out, int out_size, void* d_ws, size_t ws_size,
                              hipStream_t stream) {
  (void)in_sizes; (void)n_in; (void)out_size; (void)ws_size;
  const float* x            = (const float*)d_in[0];
  const float* lin_enc_w    = (const float*)d_in[1];
  const float* lin_enc_b    = (const float*)d_in[2];
  const float* pos_emb      = (const float*)d_in[3];
  const float* query_embed  = (const float*)d_in[4];
  const float* enc_in_w     = (const float*)d_in[5];
  const float* enc_out_w    = (const float*)d_in[6];
  const float* enc_l1_w     = (const float*)d_in[7];
  const float* enc_l1_b     = (const float*)d_in[8];
  const float* enc_l2_w     = (const float*)d_in[9];
  const float* enc_l2_b     = (const float*)d_in[10];
  const float* enc_ln1_g    = (const float*)d_in[11];
  const float* enc_ln1_b    = (const float*)d_in[12];
  const float* enc_ln2_g    = (const float*)d_in[13];
  const float* enc_ln2_b    = (const float*)d_in[14];
  const float* dec_sa_in_w  = (const float*)d_in[15];
  const float* dec_sa_out_w = (const float*)d_in[16];
  const float* dec_ca_in_w  = (const float*)d_in[17];
  const float* dec_ca_out_w = (const float*)d_in[18];
  const float* dec_l1_w     = (const float*)d_in[19];
  const float* dec_l1_b     = (const float*)d_in[20];
  const float* dec_l2_w     = (const float*)d_in[21];
  const float* dec_l2_b     = (const float*)d_in[22];
  const float* dec_ln1_g    = (const float*)d_in[23];
  const float* dec_ln1_b    = (const float*)d_in[24];
  const float* dec_ln2_g    = (const float*)d_in[25];
  const float* dec_ln2_b    = (const float*)d_in[26];
  const float* dec_ln3_g    = (const float*)d_in[27];
  const float* dec_ln3_b    = (const float*)d_in[28];
  const float* head1_w      = (const float*)d_in[29];
  const float* head1_b      = (const float*)d_in[30];
  const float* head2_w      = (const float*)d_in[31];
  const float* head2_b      = (const float*)d_in[32];
  const int*   qidx         = (const int*)d_in[33];
  float* yout = (float*)d_out;

  // --- workspace carve-up (all offsets 256B aligned) ---
  char* p = (char*)d_ws;
  auto alloc = [&](size_t bytes) -> void* {
    char* r = p;
    p += (bytes + 255) & ~(size_t)255;
    return (void*)r;
  };
  float*    partials = (float*)alloc(256 * sizeof(float));
  float*    alphaBuf = (float*)alloc(256);
  _Float16* qw       = (_Float16*)alloc((size_t)HID_ * D_ * 2);     // largest weight
  float*    bufT     = (float*)alloc((size_t)NTOK * D_ * 4);
  float*    bufH     = (float*)alloc((size_t)NTOK * D_ * 4);
  float*    bufO     = (float*)alloc((size_t)NTOK * D_ * 4);
  float*    bufLN    = (float*)alloc((size_t)NTOK * D_ * 4);
  float*    attcat   = (float*)alloc((size_t)NTOK * D_ * 4);
  float*    qkv      = (float*)alloc((size_t)NTOK * D3 * 4);
  float*    ffn32    = (float*)alloc((size_t)NTOK * HID_ * 4);
  float*    scores   = (float*)alloc((size_t)BH * S_ * S_ * 4);
  _Float16* ln16     = (_Float16*)alloc((size_t)NTOK * D_ * 2);
  _Float16* aux16    = (_Float16*)alloc((size_t)NTOK * D_ * 2);
  _Float16* cat16    = (_Float16*)alloc((size_t)NTOK * D_ * 2);
  _Float16* mem16    = (_Float16*)alloc((size_t)NTOK * D_ * 2);
  _Float16* ffn16    = (_Float16*)alloc((size_t)NTOK * HID_ * 2);
  _Float16* qp       = (_Float16*)alloc((size_t)BH * S_ * HDP * 2);
  _Float16* kp       = (_Float16*)alloc((size_t)BH * S_ * HDP * 2);
  _Float16* vt       = (_Float16*)alloc((size_t)BH * HD_ * S_ * 2);
  _Float16* att16    = (_Float16*)alloc((size_t)BH * S_ * S_ * 2);

  auto blocks = [](long long n) { return (unsigned)((n + 255) / 256); };

  auto quant = [&](const float* w, long long n) {
    abssum_k<<<256, 256, 0, stream>>>(w, n, partials);
    alpha_k<<<1, 32, 0, stream>>>(partials, 256, 1.0f / (float)n, alphaBuf);
    quant_k<<<blocks(n), 256, 0, stream>>>(w, alphaBuf, qw, n);
  };
  auto gemm = [&](const _Float16* A, const _Float16* W, const float* bias, const float* R,
                  float* C, int M, int N, int K, int lda, int ldb, int ldc, int batch,
                  long long aS, long long wS, long long cS, long long rS, int flags) {
    if (N % 64 == 0) {
      const int tiles = (M / 32) * (N / 64);
      dim3 grid((tiles + 7) / 8, batch);
      gemm_wmma_k<4><<<grid, 256, 0, stream>>>(A, W, bias, R, C, M, N, K, lda, ldb, ldc,
                                               aS, wS, cS, rS, flags);
    } else if (N % 48 == 0) {
      const int tiles = (M / 32) * (N / 48);
      dim3 grid((tiles + 7) / 8, batch);
      gemm_wmma_k<3><<<grid, 256, 0, stream>>>(A, W, bias, R, C, M, N, K, lda, ldb, ldc,
                                               aS, wS, cS, rS, flags);
    } else {
      const int tiles = (M / 32) * (N / 16);
      dim3 grid((tiles + 7) / 8, batch);
      gemm_wmma_k<1><<<grid, 256, 0, stream>>>(A, W, bias, R, C, M, N, K, lda, ldb, ldc,
                                               aS, wS, cS, rS, flags);
    }
  };
  // full MHA core given qkv buffer filled: writes (S,B,D) f32 attcat + f16 cat16
  auto attention = [&]() {
    repack_qk_k<<<blocks((long long)BH * S_ * HDP), 256, 0, stream>>>(qkv, qp, kp);
    repack_vt_k<<<blocks((long long)BH * HD_ * S_), 256, 0, stream>>>(qkv, vt);
    gemm(qp, kp, nullptr, nullptr, scores, S_, S_, HDP, HDP, HDP, S_, BH,
         (long long)S_ * HDP, (long long)S_ * HDP, (long long)S_ * S_, 0, 0);
    softmax_k<<<BH * S_, 256, 0, stream>>>(scores, att16);
    gemm(att16, vt, nullptr, nullptr, attcat, S_, HD_, S_, S_, S_, B_ * D_, BH,
         (long long)S_ * S_, (long long)HD_ * S_, (long long)HD_, 0, 0);
    cast16_k<<<blocks(NTOK * D_), 256, 0, stream>>>(attcat, cat16, NTOK * D_);
  };

  const int nTok = NTOK * D_;
  const long long wInW  = (long long)D3 * D_;    // 3D x D
  const long long wOutW = (long long)D_ * D_;    // D x D
  const long long wL1   = (long long)HID_ * D_;  // HID x D
  const long long wL2   = (long long)D_ * HID_;  // D x HID

  // ---- input projection: t = qlinear(t, lin_enc_w, lin_enc_b) + t ----
  patchify_k<<<blocks(nTok), 256, 0, stream>>>(x, bufT, ln16);
  quant(lin_enc_w, wOutW);
  gemm(ln16, qw, lin_enc_b, bufT, bufT, NTOK, D_, D_, D_, D_, D_, 1, 0, 0, 0, 0, 1 | 4);
  // h = t + pos
  addrow_k<<<blocks(nTok), 256, 0, stream>>>(bufT, pos_emb, nullptr, 0, bufH, nullptr, nTok);

  // ---- encoder ----
  for (int l = 0; l < NL_; ++l) {
    layernorm_k<<<NTOK, 256, 0, stream>>>(bufH, enc_ln1_g + l * D_, enc_ln1_b + l * D_,
                                          bufLN, ln16);
    quant(enc_in_w + l * wInW, wInW);
    gemm(ln16, qw, nullptr, nullptr, qkv, NTOK, D3, D_, D_, D_, D3, 1, 0, 0, 0, 0, 0);
    attention();
    quant(enc_out_w + l * wOutW, wOutW);
    gemm(cat16, qw, nullptr, bufH, bufH, NTOK, D_, D_, D_, D_, D_, 1, 0, 0, 0, 0, 4);

    layernorm_k<<<NTOK, 256, 0, stream>>>(bufH, enc_ln2_g + l * D_, enc_ln2_b + l * D_,
                                          bufLN, ln16);
    quant(enc_l1_w + l * wL1, wL1);
    gemm(ln16, qw, enc_l1_b + l * HID_, nullptr, ffn32, NTOK, HID_, D_, D_, D_, HID_,
         1, 0, 0, 0, 0, 1 | 2);
    cast16_k<<<blocks((long long)NTOK * HID_), 256, 0, stream>>>(ffn32, ffn16, NTOK * HID_);
    quant(enc_l2_w + l * wL2, wL2);
    gemm(ffn16, qw, enc_l2_b + l * D_, bufH, bufH, NTOK, D_, HID_, HID_, HID_, D_,
         1, 0, 0, 0, 0, 1 | 4);
  }

  // mem = h ; o = mem
  cast16_k<<<blocks(nTok), 256, 0, stream>>>(bufH, mem16, nTok);
  copy_k<<<blocks(nTok), 256, 0, stream>>>(bufH, bufO, nTok);

  // ---- decoder ----
  const long long qeScale = (long long)S_ * D_;
  for (int l = 0; l < NL_; ++l) {
    // self-attention: q = k = o2 + qe, v = o2
    layernorm_k<<<NTOK, 256, 0, stream>>>(bufO, dec_ln1_g + l * D_, dec_ln1_b + l * D_,
                                          bufLN, ln16);
    addrow_k<<<blocks(nTok), 256, 0, stream>>>(bufLN, query_embed, qidx, qeScale,
                                               nullptr, aux16, nTok);
    quant(dec_sa_in_w + l * wInW, wInW);
    gemm(aux16, qw,             nullptr, nullptr, qkv,          NTOK, D_, D_, D_, D_, D3, 1, 0, 0, 0, 0, 0);
    gemm(aux16, qw + wOutW,     nullptr, nullptr, qkv + D_,     NTOK, D_, D_, D_, D_, D3, 1, 0, 0, 0, 0, 0);
    gemm(ln16,  qw + 2 * wOutW, nullptr, nullptr, qkv + 2 * D_, NTOK, D_, D_, D_, D_, D3, 1, 0, 0, 0, 0, 0);
    attention();
    quant(dec_sa_out_w + l * wOutW, wOutW);
    gemm(cat16, qw, nullptr, bufO, bufO, NTOK, D_, D_, D_, D_, D_, 1, 0, 0, 0, 0, 4);

    // cross-attention: q = o2 + qe, k = v = mem
    layernorm_k<<<NTOK, 256, 0, stream>>>(bufO, dec_ln2_g + l * D_, dec_ln2_b + l * D_,
                                          bufLN, ln16);
    addrow_k<<<blocks(nTok), 256, 0, stream>>>(bufLN, query_embed, qidx, qeScale,
                                               nullptr, aux16, nTok);
    quant(dec_ca_in_w + l * wInW, wInW);
    gemm(aux16, qw,             nullptr, nullptr, qkv,          NTOK, D_, D_, D_, D_, D3, 1, 0, 0, 0, 0, 0);
    gemm(mem16, qw + wOutW,     nullptr, nullptr, qkv + D_,     NTOK, D_, D_, D_, D_, D3, 1, 0, 0, 0, 0, 0);
    gemm(mem16, qw + 2 * wOutW, nullptr, nullptr, qkv + 2 * D_, NTOK, D_, D_, D_, D_, D3, 1, 0, 0, 0, 0, 0);
    attention();
    quant(dec_ca_out_w + l * wOutW, wOutW);
    gemm(cat16, qw, nullptr, bufO, bufO, NTOK, D_, D_, D_, D_, D_, 1, 0, 0, 0, 0, 4);

    // FFN
    layernorm_k<<<NTOK, 256, 0, stream>>>(bufO, dec_ln3_g + l * D_, dec_ln3_b + l * D_,
                                          bufLN, ln16);
    quant(dec_l1_w + l * wL1, wL1);
    gemm(ln16, qw, dec_l1_b + l * HID_, nullptr, ffn32, NTOK, HID_, D_, D_, D_, HID_,
         1, 0, 0, 0, 0, 1 | 2);
    cast16_k<<<blocks((long long)NTOK * HID_), 256, 0, stream>>>(ffn32, ffn16, NTOK * HID_);
    quant(dec_l2_w + l * wL2, wL2);
    gemm(ffn16, qw, dec_l2_b + l * D_, bufO, bufO, NTOK, D_, HID_, HID_, HID_, D_,
         1, 0, 0, 0, 0, 1 | 4);
  }

  // ---- head: o = o + qlinear(relu(qlinear(o, head1)), head2) ----
  cast16_k<<<blocks(nTok), 256, 0, stream>>>(bufO, ln16, nTok);
  quant(head1_w, wL1);
  gemm(ln16, qw, head1_b, nullptr, ffn32, NTOK, HID_, D_, D_, D_, HID_, 1, 0, 0, 0, 0, 1 | 2);
  cast16_k<<<blocks((long long)NTOK * HID_), 256, 0, stream>>>(ffn32, ffn16, NTOK * HID_);
  quant(head2_w, wL2);
  gemm(ffn16, qw, head2_b, bufO, bufO, NTOK, D_, HID_, HID_, HID_, D_, 1, 0, 0, 0, 0, 1 | 4);

  unpatchify_k<<<blocks((long long)B_ * 64 * 48 * 48), 256, 0, stream>>>(bufO, yout);
}
